// MambaTrans_52304111730831
// MI455X (gfx1250) — compile-verified, matrix operand order checked by reading, not run on hardware
//
#include <hip/hip_runtime.h>
#include <hip/hip_bf16.h>

// ---------------------------------------------------------------------------
// Mamba block + head for gfx1250 (MI455X).
// GEMMs: v_wmma_f32_16x16x32_f16, one wave per 16x32 output tile (2 accums),
// guard-free inner loop (all dims padded to WMMA multiples), b128 fragment
// loads, global_prefetch_b8 ahead of the K-loop.
// Selective scan: one wave32 per (batch, channel); 8 states per lane in VGPRs,
// v_exp_f32 discretization, __shfl_xor wave reduction for the C.h contraction.
// ---------------------------------------------------------------------------

typedef _Float16 half_t;
typedef __attribute__((ext_vector_type(16))) _Float16 v16h;
typedef __attribute__((ext_vector_type(8)))  float    v8f;
typedef __attribute__((ext_vector_type(4)))  unsigned uint4v;   // 16B aligned

// Problem constants (match reference)
constexpr int Bn    = 2;
constexpr int Ln    = 512;
constexpr int Cn    = 384;   // d_model
constexpr int DIn   = 768;   // d_inner
constexpr int Sn    = 256;   // d_state
constexpr int Rn    = 24;    // dt_rank
constexpr int Kc    = 4;     // d_conv
constexpr int ROWS  = Bn * Ln;          // 1024
constexpr int NDBL  = Rn + 2 * Sn;      // 536 (true width)
constexpr int NDBLP = 544;              // padded to multiple of 32 for WMMA
constexpr int TWODI = 2 * DIn;          // 1536

// ---------------------------------------------------------------------------
// f32 -> f16 converter (weights)
// ---------------------------------------------------------------------------
__global__ void k_f32_to_f16(const float* __restrict__ in, half_t* __restrict__ out, int n) {
    int i = blockIdx.x * blockDim.x + threadIdx.x;
    if (i < n) out[i] = (half_t)in[i];
}

// ---------------------------------------------------------------------------
// LayerNorm over last dim. One wave32 per row; output f16 (GEMM operand).
// ---------------------------------------------------------------------------
__global__ void k_layernorm_f16(const float* __restrict__ x,
                                const float* __restrict__ g,
                                const float* __restrict__ b,
                                half_t* __restrict__ out, int rows, int cols) {
    int wid  = (blockIdx.x * blockDim.x + threadIdx.x) >> 5;
    int lane = threadIdx.x & 31;
    if (wid >= rows) return;
    const float* row = x + (size_t)wid * cols;
    float s = 0.f, s2 = 0.f;
    for (int i = lane; i < cols; i += 32) { float v = row[i]; s += v; s2 += v * v; }
#pragma unroll
    for (int m = 16; m; m >>= 1) { s += __shfl_xor(s, m, 32); s2 += __shfl_xor(s2, m, 32); }
    float mu  = s / cols;
    float var = s2 / cols - mu * mu;
    float rs  = rsqrtf(var + 1e-5f);
    half_t* orow = out + (size_t)wid * cols;
    for (int i = lane; i < cols; i += 32)
        orow[i] = (half_t)((row[i] - mu) * rs * g[i] + b[i]);
}

// ---------------------------------------------------------------------------
// WMMA GEMM: out[M,N] f32 = A[M,K] f16 @ W[N,K]^T f16  (+bias[n]) (+res[m,n])
// Requirements (guaranteed by launch): M%16==0, N%32==0, K%32==0.
// One wave per 16x32 output tile (two 16x16 WMMA accumulators sharing A).
// Per-lane fragment addressing (ISA 7.12.2, 16-bit operands):
//   A 16x32 : row = lane&15, khalf = (lane>>4)*8; dwords cover K offsets
//             {khalf..khalf+7} and {khalf+16..khalf+23} -> two b128 loads.
//   B 32x16 : same mapping with lane&15 = W row (output column).
//   C/D     : m = r + 8*(lane>>4), n = lane&15.
// ---------------------------------------------------------------------------
__global__ void k_gemm_wmma(const half_t* __restrict__ A,
                            const half_t* __restrict__ W,
                            float* __restrict__ out,
                            const float* __restrict__ bias,
                            const float* __restrict__ res,
                            int M, int N, int K) {
    int gwave = (blockIdx.x * blockDim.x + threadIdx.x) >> 5;
    int lane  = threadIdx.x & 31;
    int tilesN = N >> 5;                       // 32-wide N tiles
    int tiles  = (M >> 4) * tilesN;
    if (gwave >= tiles) return;
    int tm = gwave / tilesN;
    int tn = gwave % tilesN;

    int l15   = lane & 15;
    int khalf = (lane >> 4) * 8;

    const half_t* arow = A + (size_t)(tm * 16 + l15) * K + khalf;
    const half_t* w0   = W + (size_t)(tn * 32 + l15) * K + khalf;
    const half_t* w1   = w0 + (size_t)16 * K;

    union V16 { v16h v; uint4v q[2]; };
    v8f acc0 = {}, acc1 = {};

    for (int k0 = 0; k0 < K; k0 += 32) {
        if (k0 + 32 < K) {                     // gfx1250 global_prefetch_b8
            __builtin_prefetch(arow + k0 + 32, 0, 1);
            __builtin_prefetch(w0 + k0 + 32, 0, 1);
            __builtin_prefetch(w1 + k0 + 32, 0, 1);
        }
        V16 a, b0, b1;
        a.q[0]  = *(const uint4v*)(arow + k0);
        a.q[1]  = *(const uint4v*)(arow + k0 + 16);
        b0.q[0] = *(const uint4v*)(w0 + k0);
        b0.q[1] = *(const uint4v*)(w0 + k0 + 16);
        b1.q[0] = *(const uint4v*)(w1 + k0);
        b1.q[1] = *(const uint4v*)(w1 + k0 + 16);
        acc0 = __builtin_amdgcn_wmma_f32_16x16x32_f16(
            false, a.v, false, b0.v, (short)0, acc0, false, false);
        acc1 = __builtin_amdgcn_wmma_f32_16x16x32_f16(
            false, a.v, false, b1.v, (short)0, acc1, false, false);
    }

    int n0    = tn * 32 + l15;                 // second tile at n0+16
    int mbase = tm * 16 + ((lane >> 4) * 8);
#pragma unroll
    for (int r = 0; r < 8; ++r) {
        int m = mbase + r;
        float v0 = acc0[r];
        float v1 = acc1[r];
        if (bias) { v0 += bias[n0]; v1 += bias[n0 + 16]; }
        if (res)  { v0 += res[(size_t)m * N + n0]; v1 += res[(size_t)m * N + n0 + 16]; }
        out[(size_t)m * N + n0]      = v0;
        out[(size_t)m * N + n0 + 16] = v1;
    }
}

// ---------------------------------------------------------------------------
// Causal depthwise conv1d (K=4) + bias + SiLU.  x_in = xz[:, :, 0:DI]
// Clamped-index unconditional loads (no exec divergence).
// ---------------------------------------------------------------------------
__global__ void k_conv_silu(const float* __restrict__ xz,
                            const float* __restrict__ conv_w,
                            const float* __restrict__ conv_b,
                            float* __restrict__ x_act,
                            half_t* __restrict__ x_act_h) {
    int idx = blockIdx.x * blockDim.x + threadIdx.x;   // over B*L*DI
    if (idx >= Bn * Ln * DIn) return;
    int d = idx % DIn;
    int l = (idx / DIn) % Ln;
    int b = idx / (DIn * Ln);
    const float* base = xz + (size_t)b * Ln * TWODI;   // x_in: stride TWODI, col d
    float acc = conv_b[d];
#pragma unroll
    for (int j = 0; j < Kc; ++j) {
        int ll = l - (Kc - 1) + j;
        int lc = ll < 0 ? 0 : ll;
        float mask = ll < 0 ? 0.f : 1.f;
        acc += conv_w[d * Kc + j] * mask * base[(size_t)lc * TWODI + d];
    }
    float sv = acc / (1.f + __expf(-acc));             // silu
    x_act[idx]   = sv;
    x_act_h[idx] = (half_t)sv;
}

// ---------------------------------------------------------------------------
// dt = softplus(x_dbl[:, :R] @ dt_proj_w^T + dt_proj_b); dtu = dt * x_act
// K=24 — plain FMA loop (too small for WMMA). x_dbl has padded stride NDBLP.
// ---------------------------------------------------------------------------
__global__ void k_dt(const float* __restrict__ x_dbl,
                     const float* __restrict__ dtw,
                     const float* __restrict__ dtb,
                     const float* __restrict__ x_act,
                     float* __restrict__ dt,
                     float* __restrict__ dtu) {
    int idx = blockIdx.x * blockDim.x + threadIdx.x;   // over ROWS*DI
    if (idx >= ROWS * DIn) return;
    int d   = idx % DIn;
    int row = idx / DIn;
    const float* xr = x_dbl + (size_t)row * NDBLP;
    float acc = dtb[d];
#pragma unroll
    for (int r = 0; r < Rn; ++r) acc += xr[r] * dtw[d * Rn + r];
    float sp = (acc > 20.f) ? acc : log1pf(__expf(acc));  // softplus
    dt[idx]  = sp;
    dtu[idx] = sp * x_act[idx];
}

// ---------------------------------------------------------------------------
// Selective scan. One wave32 per (b, d); lane holds states s = lane + 32*j.
// h = exp(dt*A)*h + dtu*B_t;  y = sum_s h*C_t (wave reduction).
// ---------------------------------------------------------------------------
__global__ void k_scan(const float* __restrict__ dt,
                       const float* __restrict__ dtu,
                       const float* __restrict__ x_dbl,
                       const float* __restrict__ A_log,
                       float* __restrict__ ys) {
    int gw   = (blockIdx.x * blockDim.x + threadIdx.x) >> 5;
    int lane = threadIdx.x & 31;
    if (gw >= Bn * DIn) return;
    int b = gw / DIn;
    int d = gw % DIn;

    float A[8], h[8];
#pragma unroll
    for (int j = 0; j < 8; ++j) {
        int s = lane + 32 * j;
        A[j] = -__expf(A_log[(size_t)d * Sn + s]);
        h[j] = 0.f;
    }

    const float* xdb  = x_dbl + (size_t)b * Ln * NDBLP;
    const float* dtp  = dt  + (size_t)b * Ln * DIn + d;
    const float* dtup = dtu + (size_t)b * Ln * DIn + d;
    float*       yp   = ys  + (size_t)b * Ln * DIn + d;

    for (int l = 0; l < Ln; ++l) {
        float dtv  = dtp [(size_t)l * DIn];
        float dtuv = dtup[(size_t)l * DIn];
        const float* Brow = xdb + (size_t)l * NDBLP + Rn;
        const float* Crow = Brow + Sn;
        float acc = 0.f;
#pragma unroll
        for (int j = 0; j < 8; ++j) {
            int s = lane + 32 * j;
            float dA = __expf(dtv * A[j]);          // v_exp_f32
            h[j] = dA * h[j] + dtuv * Brow[s];
            acc += h[j] * Crow[s];
        }
#pragma unroll
        for (int m = 16; m; m >>= 1) acc += __shfl_xor(acc, m, 32);
        if (lane == 0) yp[(size_t)l * DIn] = acc;
    }
}

// ---------------------------------------------------------------------------
// y = (ys + D * x_act) * silu(z) -> f16   (z = xz[:, :, DI:2DI])
// ---------------------------------------------------------------------------
__global__ void k_gate(const float* __restrict__ ys,
                       const float* __restrict__ x_act,
                       const float* __restrict__ Dw,
                       const float* __restrict__ xz,
                       half_t* __restrict__ y_h) {
    int idx = blockIdx.x * blockDim.x + threadIdx.x;   // over ROWS*DI
    if (idx >= ROWS * DIn) return;
    int d   = idx % DIn;
    int row = idx / DIn;
    float z = xz[(size_t)row * TWODI + DIn + d];
    float y = (ys[idx] + Dw[d] * x_act[idx]) * (z / (1.f + __expf(-z)));
    y_h[idx] = (half_t)y;
}

// ---------------------------------------------------------------------------
// launch
// ---------------------------------------------------------------------------
static inline size_t align_up(size_t v, size_t a) { return (v + a - 1) & ~(a - 1); }

extern "C" void kernel_launch(void* const* d_in, const int* in_sizes, int n_in,
                              void* d_out, int out_size, void* d_ws, size_t ws_size,
                              hipStream_t stream) {
    const float* x         = (const float*)d_in[0];
    const float* ln1_g     = (const float*)d_in[1];
    const float* ln1_b     = (const float*)d_in[2];
    const float* ln2_g     = (const float*)d_in[3];
    const float* ln2_b     = (const float*)d_in[4];
    const float* head_w    = (const float*)d_in[5];
    const float* head_b    = (const float*)d_in[6];
    const float* in_proj_w = (const float*)d_in[7];
    const float* conv_w    = (const float*)d_in[8];
    const float* conv_b    = (const float*)d_in[9];
    const float* x_proj_w  = (const float*)d_in[10];
    const float* dt_proj_w = (const float*)d_in[11];
    const float* dt_proj_b = (const float*)d_in[12];
    const float* A_log     = (const float*)d_in[13];
    const float* Dw        = (const float*)d_in[14];
    const float* out_proj_w= (const float*)d_in[15];
    float* out = (float*)d_out;

    // workspace carve-up
    char* p = (char*)d_ws;
    size_t off = 0;
    auto alloc = [&](size_t bytes) { void* r = p + off; off = align_up(off + bytes, 256); return r; };

    half_t* ln1_h    = (half_t*)alloc((size_t)ROWS * Cn * 2);
    half_t* w_in_h   = (half_t*)alloc((size_t)TWODI * Cn * 2);
    float*  xz       = (float*) alloc((size_t)ROWS * TWODI * 4);
    float*  x_act    = (float*) alloc((size_t)ROWS * DIn * 4);
    half_t* x_act_h  = (half_t*)alloc((size_t)ROWS * DIn * 2);
    half_t* w_xp_h   = (half_t*)alloc((size_t)NDBLP * DIn * 2);  // rows 536..543 garbage (feed only x_dbl cols 536..543)
    float*  x_dbl    = (float*) alloc((size_t)ROWS * NDBLP * 4); // cols 536..543 garbage (never read)
    float*  dt       = (float*) alloc((size_t)ROWS * DIn * 4);
    float*  dtu      = (float*) alloc((size_t)ROWS * DIn * 4);
    float*  ys       = (float*) alloc((size_t)ROWS * DIn * 4);
    half_t* y_h      = (half_t*)alloc((size_t)ROWS * DIn * 2);
    half_t* w_out_h  = (half_t*)alloc((size_t)Cn * DIn * 2);
    float*  x1       = (float*) alloc((size_t)ROWS * Cn * 4);
    half_t* ln2_h    = (half_t*)alloc((size_t)ROWS * Cn * 2);
    half_t* w_head_h = (half_t*)alloc((size_t)Cn * Cn * 2);
    (void)ws_size;

    const int BT = 256;
    auto cdiv = [](int a, int b) { return (a + b - 1) / b; };

    // weights -> f16
    {
        int n;
        n = TWODI * Cn;  k_f32_to_f16<<<cdiv(n, BT), BT, 0, stream>>>(in_proj_w, w_in_h, n);
        n = NDBL * DIn;  k_f32_to_f16<<<cdiv(n, BT), BT, 0, stream>>>(x_proj_w,  w_xp_h, n);
        n = Cn * DIn;    k_f32_to_f16<<<cdiv(n, BT), BT, 0, stream>>>(out_proj_w, w_out_h, n);
        n = Cn * Cn;     k_f32_to_f16<<<cdiv(n, BT), BT, 0, stream>>>(head_w,    w_head_h, n);
    }

    // ln1
    k_layernorm_f16<<<cdiv(ROWS * 32, BT), BT, 0, stream>>>(x, ln1_g, ln1_b, ln1_h, ROWS, Cn);

    // in_proj: xz = ln1 @ in_proj_w^T   (1024 x 384 x 1536)
    {
        int tiles = (ROWS / 16) * (TWODI / 32);
        k_gemm_wmma<<<cdiv(tiles * 32, BT), BT, 0, stream>>>(ln1_h, w_in_h, xz,
                                                             nullptr, nullptr, ROWS, TWODI, Cn);
    }

    // conv + silu
    k_conv_silu<<<cdiv(Bn * Ln * DIn, BT), BT, 0, stream>>>(xz, conv_w, conv_b, x_act, x_act_h);

    // x_proj: x_dbl = x_act @ x_proj_w^T  (1024 x 768 x 544-padded)
    {
        int tiles = (ROWS / 16) * (NDBLP / 32);
        k_gemm_wmma<<<cdiv(tiles * 32, BT), BT, 0, stream>>>(x_act_h, w_xp_h, x_dbl,
                                                             nullptr, nullptr, ROWS, NDBLP, DIn);
    }

    // dt + dtu
    k_dt<<<cdiv(ROWS * DIn, BT), BT, 0, stream>>>(x_dbl, dt_proj_w, dt_proj_b, x_act, dt, dtu);

    // selective scan
    k_scan<<<cdiv(Bn * DIn * 32, BT), BT, 0, stream>>>(dt, dtu, x_dbl, A_log, ys);

    // gate
    k_gate<<<cdiv(ROWS * DIn, BT), BT, 0, stream>>>(ys, x_act, Dw, xz, y_h);

    // out_proj + residual x:  x1 = y @ out_proj_w^T + x   (1024 x 768 x 384)
    {
        int tiles = (ROWS / 16) * (Cn / 32);
        k_gemm_wmma<<<cdiv(tiles * 32, BT), BT, 0, stream>>>(y_h, w_out_h, x1,
                                                             nullptr, x, ROWS, Cn, DIn);
    }

    // ln2
    k_layernorm_f16<<<cdiv(ROWS * 32, BT), BT, 0, stream>>>(x1, ln2_g, ln2_b, ln2_h, ROWS, Cn);

    // head: out = ln2 @ head_w^T + head_b + x1   (1024 x 384 x 384)
    {
        int tiles = (ROWS / 16) * (Cn / 32);
        k_gemm_wmma<<<cdiv(tiles * 32, BT), BT, 0, stream>>>(ln2_h, w_head_h, out,
                                                             head_b, x1, ROWS, Cn, Cn);
    }
    (void)in_sizes; (void)n_in; (void)out_size;
}
